// MPNEncoder_58394375356586
// MI455X (gfx1250) — matrix operand order repaired and checked by reading, not compiled, not run on hardware
//
#include <hip/hip_runtime.h>
#include <hip/hip_bf16.h>

typedef __attribute__((ext_vector_type(16))) __bf16 v16bf;
typedef __attribute__((ext_vector_type(8)))  float  v8f;

union Frag {
    v16bf v;
    uint4 q[2];
};

#define NPAD 320
#define NTILES 20
#define HDIM 300

// ---------------------------------------------------------------------------
// Convert an fp32 weight matrix [Ksrc x 300] (row-major) into fragment-ordered
// bf16, zero-padded to [Kpad x 320].  Layout: for N-tile nt, K-step ks, lane l,
// element e -> Wf[((nt*nK + ks)*32 + l)*16 + e], where
//   l = 16*((k%32)>=16) + (n%16),  e = k%16   (contiguous k order, same as the A fragment)
// ---------------------------------------------------------------------------
__global__ void convert_wfrag_kernel(const float* __restrict__ W,
                                     __bf16* __restrict__ Wf,
                                     int Ksrc, int Nsrc, int Kpad)
{
    int t = blockIdx.x * blockDim.x + threadIdx.x;
    int total = Kpad * NPAD;
    if (t >= total) return;
    int kp = t / NPAD;
    int np = t - kp * NPAD;
    float v = (kp < Ksrc && np < Nsrc) ? W[(size_t)kp * Nsrc + np] : 0.0f;
    int nK = Kpad >> 5;
    int nt = np >> 4;
    int ks = kp >> 5;
    int l  = (((kp >> 4) & 1) << 4) | (np & 15);
    int e  = kp & 15;
    Wf[(((size_t)nt * nK + ks) * 32 + l) * 16 + e] = (__bf16)v;
}

// ---------------------------------------------------------------------------
// WMMA GEMM:  out[M x 300] = relu( A @ W (+ addend) (+ bias) )
//   A: fp32 [Mrows x Ksrc] (converted to bf16 during LDS fill) or
//      bf16 [Mrows x Kpad] (already zero-padded, 16B-aligned rows)
//   W: fragment-ordered bf16, Kpad multiple of 32
// Block = 256 threads (8 waves).  Block stages MT rows in LDS; each wave owns
// N-tiles nt = wave, wave+8, ... and keeps MT/16 accumulators so one B
// fragment is reused across all row tiles (cuts L2 B-traffic MT/16 x).
// ---------------------------------------------------------------------------
template<int MT, int KPMAX>
__global__ void __launch_bounds__(256)
gemm_wmma_kernel(const void* __restrict__ Asrc, int srcIsBf16, int Ksrc, int Kpad,
                 const __bf16* __restrict__ Wfrag,
                 const float* __restrict__ addend, const float* __restrict__ bias,
                 float* __restrict__ out, int Mrows)
{
    constexpr int RT = MT / 16;          // row tiles per block
    __shared__ __bf16 ldsA[MT * KPMAX];

    const int tid  = threadIdx.x;
    const int lane = tid & 31;
    const int wave = tid >> 5;
    const int m0   = blockIdx.x * MT;

    // ---- stage A tile (MT x Kpad) into LDS as bf16 ----
    if (srcIsBf16) {
        const __bf16* Ab = (const __bf16*)Asrc;
        const int kq = Kpad >> 3;        // uint4 chunks per row (Kpad mult of 8)
        for (int r = 0; r < MT; ++r) {
            int row = m0 + r;
            const uint4* src = (const uint4*)(Ab + (size_t)row * Kpad);
            uint4* dst = (uint4*)(ldsA + r * Kpad);
            uint4 z = {0u, 0u, 0u, 0u};
            for (int k = tid; k < kq; k += 256)
                dst[k] = (row < Mrows) ? src[k] : z;
        }
    } else {
        const float* Af = (const float*)Asrc;
        for (int r = 0; r < MT; ++r) {
            int row = m0 + r;
            const float* src = Af + (size_t)row * Ksrc;
            for (int k = tid; k < Kpad; k += 256) {
                float v = (row < Mrows && k < Ksrc) ? src[k] : 0.0f;
                ldsA[r * Kpad + k] = (__bf16)v;
            }
        }
    }
    __syncthreads();

    const int nK      = Kpad >> 5;
    const int halfSel = lane >> 4;       // 0: lanes 0-15, 1: lanes 16-31
    const int arow    = lane & 15;       // M index within a 16-row tile

    for (int nt = wave; nt < NTILES; nt += 8) {
        v8f acc[RT];
#pragma unroll
        for (int mt = 0; mt < RT; ++mt)
            acc[mt] = (v8f){0.f, 0.f, 0.f, 0.f, 0.f, 0.f, 0.f, 0.f};

        const __bf16* wp = Wfrag + (((size_t)nt * nK) * 32 + lane) * 16;
        for (int ks = 0; ks < nK; ++ks) {
            Frag b;                       // one B fragment reused RT times
            b.q[0] = *(const uint4*)wp;
            b.q[1] = *(const uint4*)(wp + 8);
            wp += 32 * 16;
#pragma unroll
            for (int mt = 0; mt < RT; ++mt) {
                Frag a;
                // A layout: lanes 0-15 hold K {0..15}, lanes 16-31 {16..31}
                const __bf16* ap = &ldsA[(mt * 16 + arow) * Kpad + ks * 32 + (halfSel << 4)];
                a.q[0] = *(const uint4*)ap;
                a.q[1] = *(const uint4*)(ap + 8);
                acc[mt] = __builtin_amdgcn_wmma_f32_16x16x32_bf16(
                    false, a.v, false, b.v, (short)0, acc[mt], false, false);
            }
        }
        // ---- epilogue: C/D layout -> row M = r + 8*halfSel, col N = lane&15 ----
        int col = nt * 16 + (lane & 15);
        if (col < HDIM) {
#pragma unroll
            for (int mt = 0; mt < RT; ++mt) {
                int rowBase = m0 + mt * 16 + (halfSel << 3);
#pragma unroll
                for (int r = 0; r < 8; ++r) {
                    int row = rowBase + r;
                    if (row < Mrows) {
                        float v = acc[mt][r];
                        if (addend) v += addend[(size_t)row * HDIM + col];
                        if (bias)   v += bias[col];
                        out[(size_t)row * HDIM + col] = fmaxf(v, 0.0f);
                    }
                }
            }
        }
    }
}

// ---------------------------------------------------------------------------
// a_msg[b_dst[b], :] += msg[b, :]     (float4 per thread, fp32 atomics)
// ---------------------------------------------------------------------------
__global__ void scatter_add_kernel(const float* __restrict__ msg,
                                   const int* __restrict__ dst,
                                   float* __restrict__ a_msg, int nB)
{
    int t = blockIdx.x * blockDim.x + threadIdx.x;
    int b = t / 75;               // 300/4 = 75 column groups
    if (b >= nB) return;
    int c = (t - b * 75) * 4;
    int d = dst[b];
    float4 v = *(const float4*)(msg + (size_t)b * HDIM + c);
    float* p = a_msg + (size_t)d * HDIM + c;
    unsafeAtomicAdd(p + 0, v.x);
    unsafeAtomicAdd(p + 1, v.y);
    unsafeAtomicAdd(p + 2, v.z);
    unsafeAtomicAdd(p + 3, v.w);
}

// ---------------------------------------------------------------------------
// m_bf16[b, k] = bf16( a_msg[b_src[b], k] - msg[b2revb[b], k] ), padded to 320
// ---------------------------------------------------------------------------
__global__ void compute_m_kernel(const float* __restrict__ a_msg,
                                 const float* __restrict__ msg,
                                 const int* __restrict__ b_src,
                                 const int* __restrict__ b2revb,
                                 __bf16* __restrict__ m_out, int nB)
{
    int t = blockIdx.x * blockDim.x + threadIdx.x;
    int b = t / 80;               // 320/4 = 80 column groups
    if (b >= nB) return;
    int c = (t - b * 80) * 4;
    alignas(8) __bf16 r[4];
    if (c < HDIM) {
        int s  = b_src[b];
        int rb = b2revb[b];
        float4 av = *(const float4*)(a_msg + (size_t)s  * HDIM + c);
        float4 mv = *(const float4*)(msg   + (size_t)rb * HDIM + c);
        r[0] = (__bf16)(av.x - mv.x);
        r[1] = (__bf16)(av.y - mv.y);
        r[2] = (__bf16)(av.z - mv.z);
        r[3] = (__bf16)(av.w - mv.w);
    } else {
        r[0] = r[1] = r[2] = r[3] = (__bf16)0.0f;
    }
    *(uint2*)(m_out + (size_t)b * NPAD + c) = *(uint2*)r;
}

// ---------------------------------------------------------------------------
// a_input[a, :] = bf16(concat(input_atom, a_msg, input_atom*a_msg)) pad to 928
// ---------------------------------------------------------------------------
__global__ void build_ainput_kernel(const float* __restrict__ input_atom,
                                    const float* __restrict__ a_msg,
                                    __bf16* __restrict__ out, int nA)
{
    int t = blockIdx.x * blockDim.x + threadIdx.x;
    int a = t / 232;              // 928/4 = 232 column groups
    if (a >= nA) return;
    int c = (t - a * 232) * 4;
    alignas(8) __bf16 r[4];
#pragma unroll
    for (int e = 0; e < 4; ++e) {
        int col = c + e;          // segment boundaries 300/600/900 are mult of 4
        float v;
        if (col < 300)      v = input_atom[(size_t)a * HDIM + col];
        else if (col < 600) v = a_msg[(size_t)a * HDIM + (col - 300)];
        else if (col < 900) v = input_atom[(size_t)a * HDIM + (col - 600)] *
                                a_msg[(size_t)a * HDIM + (col - 600)];
        else                v = 0.0f;
        r[e] = (__bf16)v;
    }
    *(uint2*)(out + (size_t)a * 928 + c) = *(uint2*)r;
}

// ---------------------------------------------------------------------------
// Readout: sums[mol_ids[a], :] += atom_h[a, :]
// ---------------------------------------------------------------------------
__global__ void readout_sum_kernel(const float* __restrict__ atom_h,
                                   const int* __restrict__ mol_ids,
                                   float* __restrict__ sums, int nA)
{
    int t = blockIdx.x * blockDim.x + threadIdx.x;
    int a = t / 75;
    if (a >= nA) return;
    int c = (t - a * 75) * 4;
    int m = mol_ids[a];
    float4 v = *(const float4*)(atom_h + (size_t)a * HDIM + c);
    float* p = sums + (size_t)m * HDIM + c;
    unsafeAtomicAdd(p + 0, v.x);
    unsafeAtomicAdd(p + 1, v.y);
    unsafeAtomicAdd(p + 2, v.z);
    unsafeAtomicAdd(p + 3, v.w);
}

__global__ void count_kernel(const int* __restrict__ mol_ids,
                             float* __restrict__ counts, int nA)
{
    int a = blockIdx.x * blockDim.x + threadIdx.x;
    if (a >= nA) return;
    unsafeAtomicAdd(counts + mol_ids[a], 1.0f);
}

__global__ void finalize_kernel(float* __restrict__ out,
                                const float* __restrict__ counts, int n)
{
    int i = blockIdx.x * blockDim.x + threadIdx.x;
    if (i >= n) return;
    out[i] = out[i] / fmaxf(counts[i / HDIM], 1.0f);
}

// ---------------------------------------------------------------------------
extern "C" void kernel_launch(void* const* d_in, const int* in_sizes, int n_in,
                              void* d_out, int out_size, void* d_ws, size_t ws_size,
                              hipStream_t stream)
{
    const float* f_atoms  = (const float*)d_in[0];
    const float* f_bonds  = (const float*)d_in[1];
    const float* W_i_atom = (const float*)d_in[2];
    const float* W_i_bond = (const float*)d_in[3];
    const float* W_h      = (const float*)d_in[4];
    const float* W_o      = (const float*)d_in[5];
    const float* b_o      = (const float*)d_in[6];
    const int*   b_src    = (const int*)d_in[7];
    const int*   b_dst    = (const int*)d_in[8];
    const int*   b2revb   = (const int*)d_in[9];
    const int*   mol_ids  = (const int*)d_in[10];

    const int A = 200000, B = 400000, H = 300, AF = 133, BF = 147, M = 4096;
    const int KP_I = 160, KP_H = 320, KP_O = 928;   // K padded to mult. of 32

    // ---- workspace layout ----
    char* ws = (char*)d_ws;
    size_t off = 0;
    auto alloc = [&](size_t bytes) -> void* {
        void* p = ws + off;
        off = (off + bytes + 255) & ~(size_t)255;
        return p;
    };
    __bf16* wf_ia   = (__bf16*)alloc((size_t)NTILES * (KP_I/32) * 32 * 16 * 2);
    __bf16* wf_ib   = (__bf16*)alloc((size_t)NTILES * (KP_I/32) * 32 * 16 * 2);
    __bf16* wf_h    = (__bf16*)alloc((size_t)3 * NTILES * (KP_H/32) * 32 * 16 * 2);
    __bf16* wf_o    = (__bf16*)alloc((size_t)NTILES * (KP_O/32) * 32 * 16 * 2);
    float*  counts  = (float*)alloc((size_t)M * 4);
    float*  in_atom = (float*)alloc((size_t)A * H * 4);
    float*  in_bond = (float*)alloc((size_t)B * H * 4);
    float*  message = (float*)alloc((size_t)B * H * 4);
    float*  a_msg   = (float*)alloc((size_t)A * H * 4);   // reused as atom_h
    __bf16* stage   = (__bf16*)alloc((size_t)A * 928 * 2); // m_bf16 / a_input

    auto cdiv = [](long long a, long long b) { return (int)((a + b - 1) / b); };

    // ---- weights -> fragment-ordered bf16 ----
    convert_wfrag_kernel<<<cdiv((long long)KP_I*NPAD,256),256,0,stream>>>(W_i_atom, wf_ia, AF, H, KP_I);
    convert_wfrag_kernel<<<cdiv((long long)KP_I*NPAD,256),256,0,stream>>>(W_i_bond, wf_ib, BF, H, KP_I);
    const size_t wh_layer = (size_t)NTILES * (KP_H/32) * 32 * 16;
    for (int d = 0; d < 3; ++d)
        convert_wfrag_kernel<<<cdiv((long long)KP_H*NPAD,256),256,0,stream>>>(
            W_h + (size_t)d * H * H, wf_h + d * wh_layer, H, H, KP_H);
    convert_wfrag_kernel<<<cdiv((long long)KP_O*NPAD,256),256,0,stream>>>(W_o, wf_o, 3*H, H, KP_O);

    // ---- input projections ----
    gemm_wmma_kernel<64,320><<<cdiv(A,64),256,0,stream>>>(f_atoms, 0, AF, KP_I, wf_ia,
                                                          nullptr, nullptr, in_atom, A);
    gemm_wmma_kernel<64,320><<<cdiv(B,64),256,0,stream>>>(f_bonds, 0, BF, KP_I, wf_ib,
                                                          nullptr, nullptr, in_bond, B);

    // ---- message passing (DEPTH-1 = 3 rounds) ----
    const float* msg_in = in_bond;
    for (int d = 0; d < 3; ++d) {
        hipMemsetAsync(a_msg, 0, (size_t)A * H * 4, stream);
        scatter_add_kernel<<<cdiv((long long)B*75,256),256,0,stream>>>(msg_in, b_dst, a_msg, B);
        compute_m_kernel<<<cdiv((long long)B*80,256),256,0,stream>>>(a_msg, msg_in, b_src, b2revb, stage, B);
        gemm_wmma_kernel<64,320><<<cdiv(B,64),256,0,stream>>>(stage, 1, KP_H, KP_H,
                                                              wf_h + d * wh_layer,
                                                              in_bond, nullptr, message, B);
        msg_in = message;
    }

    // ---- final atom update ----
    hipMemsetAsync(a_msg, 0, (size_t)A * H * 4, stream);
    scatter_add_kernel<<<cdiv((long long)B*75,256),256,0,stream>>>(message, b_dst, a_msg, B);
    build_ainput_kernel<<<cdiv((long long)A*232,256),256,0,stream>>>(in_atom, a_msg, stage, A);
    float* atom_h = a_msg;   // a_msg fully consumed by build_ainput -> reuse
    gemm_wmma_kernel<32,928><<<cdiv(A,32),256,0,stream>>>(stage, 1, KP_O, KP_O, wf_o,
                                                          nullptr, b_o, atom_h, A);

    // ---- mean readout per molecule ----
    hipMemsetAsync(d_out, 0, (size_t)M * H * 4, stream);
    hipMemsetAsync(counts, 0, (size_t)M * 4, stream);
    readout_sum_kernel<<<cdiv((long long)A*75,256),256,0,stream>>>(atom_h, mol_ids, (float*)d_out, A);
    count_kernel<<<cdiv(A,256),256,0,stream>>>(mol_ids, counts, A);
    finalize_kernel<<<cdiv((long long)M*H,256),256,0,stream>>>((float*)d_out, counts, M * H);
}
